// HighwayDGCNN_36541581754795
// MI455X (gfx1250) — matrix operand order repaired and checked by reading, not compiled, not run on hardware
//
#include <hip/hip_runtime.h>
#include <hip/hip_bf16.h>
#include <math.h>

// ---------------- problem constants (fixed by reference) ----------------
#define N_TOT 65536
#define NPG   1024
#define NB    64
#define K_NN  8
#define HDIM  128
#define FDIM  32
#define XCOLS 33          // FDIM + 1 (kf)
#define NEG_SLOPE 0.2f

typedef _Float16 half_t;
typedef __attribute__((ext_vector_type(2)))  _Float16 v2h;
typedef __attribute__((ext_vector_type(16))) _Float16 v16h;
typedef __attribute__((ext_vector_type(8)))  float    v8f;

#if defined(__has_builtin)
#if __has_builtin(__builtin_amdgcn_fdot2)
#define HAVE_FDOT2 1
#endif
#endif

// ---------------- WMMA helpers (gfx1250, wave32) ----------------
// A (16x32 f16): lane L holds row L&15; halves h = K koff+h (h=0..15),
// koff = (L>=16)?16:0.
// B (32x16 f16): lane L holds col L&15; halves h = K kb+h, kb=(L>=16)?16:0.
// C/D (16x16 f32): lane L, elem v -> row v + ((L>=16)?8:0), col L&15.

__device__ __forceinline__ v8f wmma_f16(v16h a, v16h b, v8f c) {
    return __builtin_amdgcn_wmma_f32_16x16x32_f16(false, a, false, b,
                                                  (short)0, c, false, false);
}

__device__ __forceinline__ v16h load_a_f32(const float* __restrict__ row,
                                           int kbase, int lane) {
    const float* rp = row + kbase + ((lane >= 16) ? 16 : 0);
    v16h a;
#pragma unroll
    for (int h = 0; h < 8; ++h) {
        a[h]     = (half_t)rp[h];
        a[h + 8] = (half_t)rp[8 + h];
    }
    return a;
}

__device__ __forceinline__ v16h load_a_h16(const half_t* __restrict__ row,
                                           int kbase, int lane) {
    const half_t* rp = row + kbase + ((lane >= 16) ? 16 : 0);
    v16h a;
#pragma unroll
    for (int h = 0; h < 8; ++h) {
        a[h]     = rp[h];
        a[h + 8] = rp[8 + h];
    }
    return a;
}

// Row-major B-fragment load (used by tiny embed kernel only).
__device__ __forceinline__ v16h load_b_lds(const half_t* __restrict__ w,
                                           int kbase, int ntbase, int lane, int ld) {
    int col = ntbase + (lane & 15);
    int kb  = kbase + ((lane >= 16) ? 16 : 0);
    v16h b;
#pragma unroll
    for (int h = 0; h < 16; ++h) b[h] = w[(kb + h) * ld + col];
    return b;
}

// ---- swizzled B-fragment storage: frag f, lane L -> 16 contiguous halves ----
// dst[((f*32 + lane)*16) + h] = W[(kc*32 + hi16 + h)][nt*16 + (lane&15)]
// with f = nt*nkc + kc, hi16 = (lane>=16)?16:0. One v16h load per fragment.
__device__ __forceinline__ void swizzle_fill(half_t* __restrict__ dst,
                                             const float* __restrict__ src,
                                             int nkc, int tid) {
    int total = nkc * 8 * 512;            // frags * (32 lanes * 16 halves)
    for (int o = tid; o < total; o += 256) {
        int f    = o >> 9;
        int lane = (o >> 4) & 31;
        int h    = o & 15;
        int kc   = f % nkc;
        int nt   = f / nkc;
        int row  = kc * 32 + ((lane >= 16) ? 16 : 0) + h;
        int col  = nt * 16 + (lane & 15);
        dst[o] = (half_t)src[row * HDIM + col];
    }
}

__device__ __forceinline__ v16h load_b_sw(const half_t* __restrict__ w,
                                          int f, int lane) {
    return *(const v16h*)(w + (((f << 5) + lane) << 4));
}

// ---------------- kernel 1: xh = relu(feat @ w0 + b0) ----------------
// grid 512 x 256 threads; one wave per 16-row tile (4096 tiles).
__global__ __launch_bounds__(256) void k_embed(const float* __restrict__ x,
                                               const float* __restrict__ w0,
                                               const float* __restrict__ b0,
                                               float* __restrict__ xh) {
    __shared__ half_t sw[FDIM * HDIM];
    __shared__ float  sb[HDIM];
    int tid = threadIdx.x;
    for (int i = tid; i < FDIM * HDIM; i += 256) sw[i] = (half_t)w0[i];
    for (int i = tid; i < HDIM; i += 256) sb[i] = b0[i];
    __syncthreads();

    int wave = tid >> 5, lane = tid & 31;
    int hi8  = (lane >= 16) ? 8 : 0;
    int mtile = blockIdx.x * 8 + wave;
    int row   = mtile * 16 + (lane & 15);
    v16h a = load_a_f32(x + (size_t)row * XCOLS, 0, lane);

#pragma unroll
    for (int nt = 0; nt < 8; ++nt) {
        v16h b = load_b_lds(sw, 0, nt * 16, lane, HDIM);
        v8f acc = {0.f, 0.f, 0.f, 0.f, 0.f, 0.f, 0.f, 0.f};
        acc = wmma_f16(a, b, acc);
        int col   = nt * 16 + (lane & 15);
        float bia = sb[col];
#pragma unroll
        for (int v = 0; v < 8; ++v) {
            int r = mtile * 16 + v + hi8;
            float val = acc[v] + bia;
            xh[(size_t)r * HDIM + col] = val > 0.f ? val : 0.f;
        }
    }
}

// ---------------- kernel 2: per-graph kNN (top-8 smallest d2) ----------------
// 1 block per graph; whole graph (1024x128 f16 = 256KB) resident in LDS.
__global__ __launch_bounds__(256) void k_knn(const float* __restrict__ xh,
                                             int* __restrict__ nbr) {
    extern __shared__ char smem[];
    half_t* sx  = (half_t*)smem;                            // NPG*HDIM halves
    float*  ssq = (float*)(smem + (size_t)NPG * HDIM * sizeof(half_t));
    int g  = blockIdx.x;
    int gb = g * NPG;
    int tid = threadIdx.x;

    for (int i = tid; i < NPG * HDIM; i += 256)
        sx[i] = (half_t)xh[(size_t)gb * HDIM + i];
    __syncthreads();
    for (int r = tid; r < NPG; r += 256) {
        const half_t* rp = sx + r * HDIM;
        float s = 0.f;
        for (int c = 0; c < HDIM; ++c) { float v = (float)rp[c]; s += v * v; }
        ssq[r] = s;
    }
    __syncthreads();

    for (int i = tid; i < NPG; i += 256) {
        const uint32_t* xi = (const uint32_t*)(sx + i * HDIM);
        uint32_t xr[HDIM / 2];
#pragma unroll
        for (int p = 0; p < HDIM / 2; ++p) xr[p] = xi[p];
        float sqi = ssq[i];
        float bd[K_NN]; int bi[K_NN];
#pragma unroll
        for (int k = 0; k < K_NN; ++k) { bd[k] = 3.0e38f; bi[k] = 0; }

        for (int j = 0; j < NPG; ++j) {
            if (j == i) continue;                       // diag += 1e9 in ref
            const uint32_t* xj = (const uint32_t*)(sx + j * HDIM);
            float dot = 0.f;
#pragma unroll 16
            for (int p = 0; p < HDIM / 2; ++p) {
                union { uint32_t u; v2h h; } ua, ub;
                ua.u = xr[p]; ub.u = xj[p];
#ifdef HAVE_FDOT2
                dot = __builtin_amdgcn_fdot2(ua.h, ub.h, dot, false);
#else
                dot += (float)ua.h[0] * (float)ub.h[0];
                dot += (float)ua.h[1] * (float)ub.h[1];
#endif
            }
            float d2 = sqi + ssq[j] - 2.f * dot;
            if (d2 < bd[K_NN - 1]) {
                // fully-unrolled stable sorted insert (static indices only)
#pragma unroll
                for (int k = K_NN - 1; k >= 0; --k) {
                    bool ck   = d2 < bd[k];
                    bool ckm1 = (k > 0) ? (d2 < bd[k - 1]) : false;
                    if (ckm1)      { bd[k] = bd[k - 1]; bi[k] = bi[k - 1]; }
                    else if (ck)   { bd[k] = d2;        bi[k] = j; }
                }
            }
        }
#pragma unroll
        for (int k = 0; k < K_NN; ++k)
            nbr[(size_t)(gb + i) * K_NN + k] = gb + bi[k];
    }
}

// ---------------- kernel 3: highway edge conv ----------------
// One wave per 16-edge tile (2 nodes x 8 neighbors). Weights live in LDS
// pre-swizzled into B-fragment order (one 32B v16h read per WMMA).
// grid 256 x 256 threads -> each wave processes 16 tiles (uniform).
__global__ __launch_bounds__(256) void k_conv(const float* __restrict__ xprev,
                                              const float* __restrict__ x,
                                              const int* __restrict__ nbr,
                                              const float* __restrict__ mw1,
                                              const float* __restrict__ mb1,
                                              const float* __restrict__ mw2,
                                              const float* __restrict__ mb2,
                                              const float* __restrict__ gw,
                                              const float* __restrict__ gb,
                                              float* __restrict__ xout) {
    extern __shared__ char smem[];
    half_t* sB1 = (half_t*)smem;                // mw1 rows 0..255 swizzled: 64 frags * 512
    half_t* sG1 = sB1 + 64 * 512;               // gw  rows 0..255 swizzled: 64 frags * 512
    half_t* sB2 = sG1 + 64 * 512;               // mw2 swizzled:             32 frags * 512
    half_t* sR1 = sB2 + 32 * 512;               // mw1 row 256 (rank-1), 128
    half_t* sRG = sR1 + HDIM;                   // gw  row 256 (rank-1), 128
    half_t* hsc = sRG + HDIM;                   // 8 waves * 16*128 h-scratch
    int tid = threadIdx.x;
    swizzle_fill(sB1, mw1, 8, tid);
    swizzle_fill(sG1, gw,  8, tid);
    swizzle_fill(sB2, mw2, 4, tid);
    for (int i = tid; i < HDIM; i += 256) {
        sR1[i] = (half_t)mw1[256 * HDIM + i];
        sRG[i] = (half_t)gw[256 * HDIM + i];
    }
    __syncthreads();

    int wave = tid >> 5, lane = tid & 31;
    int hi8  = (lane >= 16) ? 8 : 0;
    int lcol = lane & 15;
    half_t* myh = hsc + wave * 16 * HDIM;

    const int tilesPerBlock = (N_TOT / 2) / 256;     // 128
    for (int it = 0; it < tilesPerBlock / 8; ++it) { // 16 uniform iterations
        int t = blockIdx.x * tilesPerBlock + it * 8 + wave;
        // lane L owns edge row r = L&15 of this tile
        int r    = lcol;
        int node = 2 * t + (r >> 3);
        int k    = r & 7;
        int j    = nbr[(size_t)node * K_NN + k];
        float fd = x[(size_t)node * XCOLS + FDIM] - x[(size_t)j * XCOLS + FDIM];

        // A fragments of c = [xi | xj] (16 x 256), col 256 handled rank-1
        v16h afrag[8];
        const float* xi_p = xprev + (size_t)node * HDIM;
        const float* xj_p = xprev + (size_t)j * HDIM;
#pragma unroll
        for (int kc = 0; kc < 4; ++kc) afrag[kc]     = load_a_f32(xi_p, kc * 32, lane);
#pragma unroll
        for (int kc = 0; kc < 4; ++kc) afrag[kc + 4] = load_a_f32(xj_p, kc * 32, lane);

        // ---- stage 1: h = relu(c @ mw1 + mb1) -> f16 LDS scratch ----
#pragma unroll
        for (int nt = 0; nt < 8; ++nt) {
            int col = nt * 16 + lcol;
            v8f acc = {0.f, 0.f, 0.f, 0.f, 0.f, 0.f, 0.f, 0.f};
#pragma unroll
            for (int kc = 0; kc < 8; ++kc)
                acc = wmma_f16(afrag[kc], load_b_sw(sB1, nt * 8 + kc, lane), acc);
            float w256 = (float)sR1[col];
            float bia  = mb1[col];
#pragma unroll
            for (int v = 0; v < 8; ++v) {
                float fdv = __shfl(fd, v + hi8, 32);
                float val = acc[v] + fdv * w256 + bia;
                myh[(v + hi8) * HDIM + col] = (half_t)(val > 0.f ? val : 0.f);
            }
        }
        __syncthreads();   // uniform; also orders LDS h writes before reads

        v16h hfrag[4];
        {
            const half_t* hrow = myh + r * HDIM;
#pragma unroll
            for (int kc = 0; kc < 4; ++kc) hfrag[kc] = load_a_h16(hrow, kc * 32, lane);
        }

        // ---- stage 2: hout = h@mw2+mb2 ; g = sigmoid(c@gw+gb); combine ----
#pragma unroll
        for (int nt = 0; nt < 8; ++nt) {
            int col = nt * 16 + lcol;
            v8f acch = {0.f, 0.f, 0.f, 0.f, 0.f, 0.f, 0.f, 0.f};
#pragma unroll
            for (int kc = 0; kc < 4; ++kc)
                acch = wmma_f16(hfrag[kc], load_b_sw(sB2, nt * 4 + kc, lane), acch);
            v8f accg = {0.f, 0.f, 0.f, 0.f, 0.f, 0.f, 0.f, 0.f};
#pragma unroll
            for (int kc = 0; kc < 8; ++kc)
                accg = wmma_f16(afrag[kc], load_b_sw(sG1, nt * 8 + kc, lane), accg);
            float g256 = (float)sRG[col];
            float bh   = mb2[col];
            float bgv  = gb[col];
            float msum = 0.f;
#pragma unroll
            for (int v = 0; v < 8; ++v) {
                float fdv = __shfl(fd, v + hi8, 32);
                int   jv  = __shfl(j,  v + hi8, 32);
                float hval = acch[v] + bh;
                float gval = accg[v] + fdv * g256 + bgv;
                gval = 1.f / (1.f + expf(-gval));
                float xjv = xprev[(size_t)jv * HDIM + col];
                msum += gval * hval + (1.f - gval) * xjv;
            }
            // lanes 0-15 hold rows 0..7 (node 2t); lanes 16-31 rows 8..15 (node 2t+1)
            int mynode = 2 * t + (hi8 ? 1 : 0);
            float res = msum * 0.125f + xprev[(size_t)mynode * HDIM + col];
            res = res > 0.f ? res : NEG_SLOPE * res;
            xout[(size_t)mynode * HDIM + col] = res;
        }
    }
}

// ---------------- kernel 4: per-graph max pool of concat(x1,x2) ----------------
__global__ __launch_bounds__(256) void k_pool(const float* __restrict__ x1,
                                              const float* __restrict__ x2,
                                              float* __restrict__ xc) {
    int g = blockIdx.x;
    int c = threadIdx.x;   // 0..255
    const float* src = (c < HDIM) ? (x1 + (size_t)g * NPG * HDIM + c)
                                  : (x2 + (size_t)g * NPG * HDIM + (c - HDIM));
    float m = -3.4e38f;
    for (int n = 0; n < NPG; ++n) {
        float v = src[(size_t)n * HDIM];
        m = v > m ? v : m;
    }
    xc[g * 256 + c] = m;
}

// ---------------- kernel 5: classifier + log_softmax ----------------
__global__ __launch_bounds__(128) void k_cls(const float* __restrict__ xc,
                                             const float* __restrict__ cw1,
                                             const float* __restrict__ cb1,
                                             const float* __restrict__ cw2,
                                             const float* __restrict__ cb2,
                                             float* __restrict__ out) {
    __shared__ float sh[HDIM];
    __shared__ float sl[2];
    int b = blockIdx.x, t = threadIdx.x;
    const float* row = xc + b * 256;
    float acc = cb1[t];
    for (int c = 0; c < 256; ++c) acc += row[c] * cw1[c * HDIM + t];
    sh[t] = acc > 0.f ? acc : 0.f;
    __syncthreads();
    if (t < 2) {
        float l = cb2[t];
        for (int k2 = 0; k2 < HDIM; ++k2) l += sh[k2] * cw2[k2 * 2 + t];
        sl[t] = l;
    }
    __syncthreads();
    if (t == 0) {
        float l0 = sl[0], l1 = sl[1];
        float m = l0 > l1 ? l0 : l1;
        float lse = m + logf(expf(l0 - m) + expf(l1 - m));
        out[b * 2 + 0] = l0 - lse;
        out[b * 2 + 1] = l1 - lse;
    }
}

// ---------------- host launcher ----------------
extern "C" void kernel_launch(void* const* d_in, const int* in_sizes, int n_in,
                              void* d_out, int out_size, void* d_ws, size_t ws_size,
                              hipStream_t stream) {
    (void)in_sizes; (void)n_in; (void)out_size; (void)ws_size;
    const float* x      = (const float*)d_in[0];
    // d_in[1] = batch (unused: deterministic arange // NPG)
    const float* w0     = (const float*)d_in[2];
    const float* b0     = (const float*)d_in[3];
    const float* c1_mw1 = (const float*)d_in[4];
    const float* c1_mb1 = (const float*)d_in[5];
    const float* c1_mw2 = (const float*)d_in[6];
    const float* c1_mb2 = (const float*)d_in[7];
    const float* c1_gw  = (const float*)d_in[8];
    const float* c1_gb  = (const float*)d_in[9];
    const float* c2_mw1 = (const float*)d_in[10];
    const float* c2_mb1 = (const float*)d_in[11];
    const float* c2_mw2 = (const float*)d_in[12];
    const float* c2_mb2 = (const float*)d_in[13];
    const float* c2_gw  = (const float*)d_in[14];
    const float* c2_gb  = (const float*)d_in[15];
    const float* cw1    = (const float*)d_in[16];
    const float* cb1    = (const float*)d_in[17];
    const float* cw2    = (const float*)d_in[18];
    const float* cb2    = (const float*)d_in[19];

    char* ws = (char*)d_ws;
    float* xh  = (float*)ws;                         ws += (size_t)N_TOT * HDIM * 4;
    float* x1  = (float*)ws;                         ws += (size_t)N_TOT * HDIM * 4;
    float* x2  = (float*)ws;                         ws += (size_t)N_TOT * HDIM * 4;
    int*   nbr = (int*)ws;                           ws += (size_t)N_TOT * K_NN * 4;
    float* xc  = (float*)ws;

    const int KNN_LDS  = NPG * HDIM * (int)sizeof(half_t) + NPG * (int)sizeof(float); // 266240
    // swizzled weights (64+64+32 frags * 512) + 2*128 rank-1 rows + 8*2048 h-scratch
    const int CONV_LDS = (160 * 512 + 2 * HDIM + 8 * 16 * HDIM) * (int)sizeof(half_t); // 197120
    hipFuncSetAttribute((const void*)k_knn,  hipFuncAttributeMaxDynamicSharedMemorySize, KNN_LDS);
    hipFuncSetAttribute((const void*)k_conv, hipFuncAttributeMaxDynamicSharedMemorySize, CONV_LDS);

    k_embed<<<512, 256, 0, stream>>>(x, w0, b0, xh);
    k_knn<<<NB, 256, KNN_LDS, stream>>>(xh, nbr);
    k_conv<<<256, 256, CONV_LDS, stream>>>(xh, x, nbr,
                                           c1_mw1, c1_mb1, c1_mw2, c1_mb2, c1_gw, c1_gb, x1);
    k_conv<<<256, 256, CONV_LDS, stream>>>(x1, x, nbr,
                                           c2_mw1, c2_mb1, c2_mw2, c2_mb2, c2_gw, c2_gb, x2);
    k_pool<<<NB, 256, 0, stream>>>(x1, x2, xc);
    k_cls<<<NB, HDIM, 0, stream>>>(xc, cw1, cb1, cw2, cb2, (float*)d_out);
}
